// PerformerConcentrationModel_69346541961407
// MI455X (gfx1250) — matrix operand order repaired and checked by reading, not compiled, not run on hardware
//
#include <hip/hip_runtime.h>

// ---------------------------------------------------------------------------
// Performer (FAVOR+) encoder forward for MI455X (gfx1250, wave32, WMMA).
// All global tensors fp32; GEMMs convert tiles to f16 in LDS and accumulate
// fp32 via v_wmma_f32_16x16x32_f16. Staging is branchless (clamped addresses
// + cndmask zero-fill) and register-pipelined across K tiles.
// ---------------------------------------------------------------------------

typedef __attribute__((ext_vector_type(16))) _Float16 v16h;
typedef __attribute__((ext_vector_type(8)))  float    v8f;
typedef __attribute__((ext_vector_type(2)))  float    f2v;

union Frag16 { unsigned u[8]; v16h v; };
union PackH  { _Float16 h[2]; unsigned u; };

// ------------------------------ constants ----------------------------------
#define BB      256
#define NN      300
#define INDIM   1280
#define DIMC    256
#define HEADS   4
#define DH      64
#define MFEAT   266           // nb features
#define MP      272           // MFEAT padded to 16
#define NPAD    304           // N padded to 16 (kpT inner stride)
#define FFHID   1024
#define ROWS    (BB*NN)       // 76800
#define ZB      (BB*HEADS)    // 1024
#define DDZ     (NN*MP)       // 81600 floats per (b,h) dd matrix
#define KPTZ    (MP*NPAD)     // 82688
#define CTXZ    (MP*DH)       // 17408
#define LN_EPS   1e-5f
#define KEPS     1e-4f
#define QK_NORM  0.35355339059327373f   // 64^-0.25
#define DIAG_C   0.0625f                // 0.5 * 64^-0.5
#define RATIO    0.06131393394849658f   // 266^-0.5

// ---------------------------------------------------------------------------
// Generic batched WMMA GEMM:  C[z] = act(A[z] @ B[z] + bias) (+res) (*rowscale)
// Tile 128(M) x 64(N) x 32(K); 8 waves, each wave a 32x32 sub-tile (4 accs).
// Requires K even (all call sites satisfy this).
// Batch offset: z -> (z/div)*s1 + (z%div)*s2 for each operand.
// ---------------------------------------------------------------------------
__global__ __launch_bounds__(256)
void wmma_gemm_kernel(const float* __restrict__ Abase, const float* __restrict__ Bbase,
                      float* __restrict__ Cbase,
                      const float* __restrict__ biasp, const float* __restrict__ resp,
                      const float* __restrict__ rsp,
                      int M, int N, int K, int lda, int ldb, int ldc,
                      int div, int act,
                      long long sA1, long long sA2, long long sB1, long long sB2,
                      long long sC1, long long sC2, long long sR1, long long sR2,
                      long long sS1, long long sS2)
{
    __shared__ unsigned lsA[128 * 18];   // 128 rows x 16 k-pairs, stride 18 words
    __shared__ unsigned lsB[16 * 68];    // 16 k-pairs x 64 cols, stride 68 words

    const int tid  = threadIdx.x;
    const int lane = tid & 31;
    const int wave = tid >> 5;
    const int wm   = wave & 3;     // M sub-tile (0..3)
    const int wn   = wave >> 2;    // N sub-tile (0..1)
    const int lh   = lane & 15;
    const int hh   = lane >> 4;

    const int bm0 = blockIdx.y * 128;
    const int bn0 = blockIdx.x * 64;
    const int z   = blockIdx.z;
    const int zq  = z / div;
    const int zr  = z % div;

    const float* A  = Abase + (long long)zq * sA1 + (long long)zr * sA2;
    const float* Bm = Bbase + (long long)zq * sB1 + (long long)zr * sB2;
    float*       C  = Cbase + (long long)zq * sC1 + (long long)zr * sC2;
    const float* Rp = resp ? resp + (long long)zq * sR1 + (long long)zr * sR2 : nullptr;
    const float* Sp = rsp  ? rsp  + (long long)zq * sS1 + (long long)zr * sS2 : nullptr;

    v8f acc[2][2];
#pragma unroll
    for (int i = 0; i < 2; ++i)
#pragma unroll
        for (int j = 0; j < 2; ++j)
#pragma unroll
            for (int r = 0; r < 8; ++r) acc[i][j][r] = 0.f;

    // Register staging buffers (double-buffered across K tiles via pipeline).
    float ar[8][2];
    float br[4][2];

    // Branchless guarded tile load: clamp address in-bounds, zero-select OOB.
    auto load_tile = [&](int k0) {
#pragma unroll
        for (int it = 0; it < 8; ++it) {
            int idx = tid + it * 256;         // 0..2047
            int m   = idx >> 4;
            int kp  = idx & 15;
            int gm  = bm0 + m;
            int gk  = k0 + (kp << 1);
            int cm  = gm < M ? gm : M - 1;
            int ck  = gk < K ? gk : K - 2;    // K even, gk even -> ck even
            f2v f   = *(const f2v*)(A + (long long)cm * lda + ck);
            bool p  = (gm < M) & (gk < K);    // gk+1 < K iff gk < K (both even)
            ar[it][0] = p ? f.x : 0.f;
            ar[it][1] = p ? f.y : 0.f;
        }
#pragma unroll
        for (int it = 0; it < 4; ++it) {
            int idx = tid + it * 256;         // 0..1023
            int kp  = idx >> 6;
            int n   = idx & 63;
            int gn  = bn0 + n;
            int gk  = k0 + (kp << 1);
            int cn  = gn < N ? gn : N - 1;
            int ck  = gk < K ? gk : K - 2;
            const float* bp = Bm + (long long)ck * ldb + cn;
            float f0 = bp[0];
            float f1 = bp[ldb];
            bool p  = (gn < N) & (gk < K);
            br[it][0] = p ? f0 : 0.f;
            br[it][1] = p ? f1 : 0.f;
        }
    };

    auto store_tile = [&]() {
#pragma unroll
        for (int it = 0; it < 8; ++it) {
            int idx = tid + it * 256;
            int m   = idx >> 4;
            int kp  = idx & 15;
            PackH p; p.h[0] = (_Float16)ar[it][0]; p.h[1] = (_Float16)ar[it][1];
            lsA[m * 18 + kp] = p.u;
        }
#pragma unroll
        for (int it = 0; it < 4; ++it) {
            int idx = tid + it * 256;
            int kp  = idx >> 6;
            int n   = idx & 63;
            PackH p; p.h[0] = (_Float16)br[it][0]; p.h[1] = (_Float16)br[it][1];
            lsB[kp * 68 + n] = p.u;
        }
    };

    const int ktiles = (K + 31) >> 5;
    load_tile(0);
    for (int kt = 0; kt < ktiles; ++kt) {
        store_tile();
        __syncthreads();

        // ---- build fragments per ISA 7.12.2 layouts ----
        Frag16 a0, a1, b0, b1;
        const int ra0 = (wm * 32 + lh) * 18;
        const int ra1 = (wm * 32 + 16 + lh) * 18;
        const int kh2 = hh * 4;
#pragma unroll
        for (int v = 0; v < 4; ++v) {
            a0.u[v]     = lsA[ra0 + kh2 + v];
            a0.u[4 + v] = lsA[ra0 + 8 + kh2 + v];
            a1.u[v]     = lsA[ra1 + kh2 + v];
            a1.u[4 + v] = lsA[ra1 + 8 + kh2 + v];
        }
        const int c0 = wn * 32 + lh;
#pragma unroll
        for (int v = 0; v < 8; ++v) {
            b0.u[v] = lsB[(hh * 4 + (v & 3) + ((v >> 2) << 3)) * 68 + c0];
            b1.u[v] = lsB[(hh * 4 + (v & 3) + ((v >> 2) << 3)) * 68 + c0 + 16];
        }

        // kick off next tile's global loads; they overlap WMMA + next store
        if (kt + 1 < ktiles) load_tile((kt + 1) << 5);
        __syncthreads();

        acc[0][0] = __builtin_amdgcn_wmma_f32_16x16x32_f16(false, a0.v, false, b0.v, (short)0, acc[0][0], false, false);
        acc[0][1] = __builtin_amdgcn_wmma_f32_16x16x32_f16(false, a0.v, false, b1.v, (short)0, acc[0][1], false, false);
        acc[1][0] = __builtin_amdgcn_wmma_f32_16x16x32_f16(false, a1.v, false, b0.v, (short)0, acc[1][0], false, false);
        acc[1][1] = __builtin_amdgcn_wmma_f32_16x16x32_f16(false, a1.v, false, b1.v, (short)0, acc[1][1], false, false);
    }

    // ---- epilogue: bias -> act -> residual -> rowscale -> store ----
#pragma unroll
    for (int i = 0; i < 2; ++i) {
#pragma unroll
        for (int j = 0; j < 2; ++j) {
            int gcol = bn0 + wn * 32 + j * 16 + lh;
            int rowb = bm0 + wm * 32 + i * 16 + hh * 8;
            if (gcol < N) {
                float bv = biasp ? biasp[gcol] : 0.f;
#pragma unroll
                for (int r = 0; r < 8; ++r) {
                    int gm = rowb + r;
                    if (gm < M) {
                        float v = acc[i][j][r] + bv;
                        if (act == 1) {        // tanh-GELU (jax default)
                            float x = v;
                            v = 0.5f * x * (1.f + tanhf(0.7978845608028654f * (x + 0.044715f * x * x * x)));
                        } else if (act == 2) { // ReLU
                            v = fmaxf(v, 0.f);
                        }
                        if (Rp) v += Rp[(long long)gm * ldc + gcol];
                        if (Sp) v *= Sp[gm];
                        C[(long long)gm * ldc + gcol] = v;
                    }
                }
            }
        }
    }
}

// ------------------------- LayerNorm over 256-wide rows --------------------
__global__ __launch_bounds__(256)
void ln_kernel(const float* __restrict__ x, float* __restrict__ y,
               const float* __restrict__ g, const float* __restrict__ b)
{
    __shared__ float s1[256], s2[256];
    const int row = blockIdx.x, tid = threadIdx.x;
    float v = x[(long long)row * 256 + tid];
    s1[tid] = v; s2[tid] = v * v;
    __syncthreads();
    for (int s = 128; s > 0; s >>= 1) {
        if (tid < s) { s1[tid] += s1[tid + s]; s2[tid] += s2[tid + s]; }
        __syncthreads();
    }
    float mu  = s1[0] * (1.f / 256.f);
    float var = s2[0] * (1.f / 256.f) - mu * mu;
    y[(long long)row * 256 + tid] = (v - mu) * rsqrtf(var + LN_EPS) * g[tid] + b[tid];
}

// ------------- diag[z*N+n] = 0.5*norm^2 * ||x[b,n,h*64:...]||^2 ------------
__global__ __launch_bounds__(256)
void diag_kernel(const float* __restrict__ qk, float* __restrict__ diag)
{
    const int lane = threadIdx.x & 31;
    const long long row = (long long)blockIdx.x * 8 + (threadIdx.x >> 5);
    if (row >= (long long)ZB * NN) return;
    int z = (int)(row / NN), n = (int)(row % NN);
    int b = z / HEADS, h = z % HEADS;
    const float* p = qk + (long long)b * NN * DIMC + (long long)n * DIMC + h * DH;
    float a0 = p[lane * 2], a1 = p[lane * 2 + 1];
    float s = a0 * a0 + a1 * a1;
    for (int o = 16; o > 0; o >>= 1) s += __shfl_xor(s, o, 32);
    if (lane == 0) diag[row] = DIAG_C * s;
}

// -------- query features: per row, stab = rowmax; in-place exp on ddq ------
__global__ __launch_bounds__(256)
void qfeat_kernel(float* __restrict__ dd, const float* __restrict__ diag)
{
    __shared__ float red[256];
    const long long row = blockIdx.x;          // z*300 + n
    float* base = dd + row * MP;
    const int tid = threadIdx.x;
    float mx = -3.4e38f;
    for (int m = tid; m < MFEAT; m += 256) mx = fmaxf(mx, base[m]);
    red[tid] = mx; __syncthreads();
    for (int s = 128; s > 0; s >>= 1) {
        if (tid < s) red[tid] = fmaxf(red[tid], red[tid + s]);
        __syncthreads();
    }
    float stab = red[0], dg = diag[row];
    for (int m = tid; m < MFEAT; m += 256)
        base[m] = RATIO * (expf(base[m] - dg - stab) + KEPS);
}

// -------- key stabilizer: kstab[z] = max over (n<300, m<266) of ddk --------
__global__ __launch_bounds__(256)
void kstab_kernel(const float* __restrict__ ddk, float* __restrict__ kstab)
{
    __shared__ float red[256];
    const int z = blockIdx.x, tid = threadIdx.x;
    const float* base = ddk + (long long)z * DDZ;
    float mx = -3.4e38f;
    for (int i = tid; i < NN * MFEAT; i += 256) {
        int n = i / MFEAT, m = i % MFEAT;
        mx = fmaxf(mx, base[n * MP + m]);
    }
    red[tid] = mx; __syncthreads();
    for (int s = 128; s > 0; s >>= 1) {
        if (tid < s) red[tid] = fmaxf(red[tid], red[tid + s]);
        __syncthreads();
    }
    if (tid == 0) kstab[z] = red[0];
}

// ---- key features + transpose: kpT[z][m][n] = ratio*(exp(ddk-diag-stab)+eps)
__global__ __launch_bounds__(256)
void ktrans_kernel(const float* __restrict__ ddk, const float* __restrict__ diagk,
                   const float* __restrict__ kstab, float* __restrict__ kpT)
{
    __shared__ float tile[32][33];
    const int z  = blockIdx.x;
    const int n0 = blockIdx.y * 32;
    const int m0 = blockIdx.z * 32;
    const int tid = threadIdx.x;
    const float st = kstab[z];
    const float* base = ddk + (long long)z * DDZ;
    const float* dgp  = diagk + (long long)z * NN;
#pragma unroll
    for (int it = 0; it < 4; ++it) {
        int idx = tid + it * 256;
        int r = idx >> 5, c = idx & 31;       // r: n-off, c: m-off (coalesced in m)
        int gn = n0 + r, gm = m0 + c;
        float v = 0.f;
        if (gn < NN && gm < MFEAT)
            v = RATIO * (expf(base[gn * MP + gm] - dgp[gn] - st) + KEPS);
        tile[r][c] = v;
    }
    __syncthreads();
#pragma unroll
    for (int it = 0; it < 4; ++it) {
        int idx = tid + it * 256;
        int rw = idx >> 5, cw = idx & 31;     // rw: m-off, cw: n-off (coalesced in n)
        int gm = m0 + rw, gn = n0 + cw;
        if (gm < MFEAT && gn < NN)
            kpT[(long long)z * KPTZ + (long long)gm * NPAD + gn] = tile[cw][rw];
    }
}

// ------------------- ksum[z][m] = sum_n kpT[z][m][n] -----------------------
__global__ __launch_bounds__(128)
void ksum_kernel(const float* __restrict__ kpT, float* __restrict__ ksum)
{
    __shared__ float red[128];
    const int z = blockIdx.x, m = blockIdx.y, tid = threadIdx.x;
    const float* row = kpT + (long long)z * KPTZ + (long long)m * NPAD;
    float s = 0.f;
    for (int n = tid; n < NN; n += 128) s += row[n];
    red[tid] = s; __syncthreads();
    for (int w = 64; w > 0; w >>= 1) {
        if (tid < w) red[tid] += red[tid + w];
        __syncthreads();
    }
    if (tid == 0) ksum[(long long)z * MP + m] = red[0];
}

// ------------- dinv[z*300+n] = 1 / dot(qp[z,n,:], ksum[z,:]) ---------------
__global__ __launch_bounds__(128)
void dinv_kernel(const float* __restrict__ qp, const float* __restrict__ ksum,
                 float* __restrict__ dinv)
{
    __shared__ float red[128];
    const long long row = blockIdx.x;          // z*300 + n
    const int z = (int)(row / NN), tid = threadIdx.x;
    const float* qr = qp + row * MP;
    const float* ks = ksum + (long long)z * MP;
    float s = 0.f;
    for (int m = tid; m < MFEAT; m += 128) s += qr[m] * ks[m];
    red[tid] = s; __syncthreads();
    for (int w = 64; w > 0; w >>= 1) {
        if (tid < w) red[tid] += red[tid + w];
        __syncthreads();
    }
    if (tid == 0) dinv[row] = 1.f / red[0];
}

// ---- projT[d][m] = proj[m][d] * norm (per layer), ld = MP -----------------
__global__ __launch_bounds__(256)
void projT_kernel(const float* __restrict__ proj, float* __restrict__ projT)
{
    int i = blockIdx.x * 256 + threadIdx.x;
    if (i < DH * MFEAT) {
        int d = i / MFEAT, m = i % MFEAT;
        projT[d * MP + m] = proj[m * DH + d] * QK_NORM;
    }
}

// ------------------- pooled[b][c] = mean_n h[b][n][c] ----------------------
__global__ __launch_bounds__(256)
void pool_kernel(const float* __restrict__ h, float* __restrict__ pooled)
{
    const int b = blockIdx.x, c = threadIdx.x;
    float s = 0.f;
    const float* p = h + (long long)b * NN * DIMC + c;
    for (int n = 0; n < NN; ++n) s += p[(long long)n * DIMC];
    pooled[b * DIMC + c] = s * (1.f / (float)NN);
}

// --------------- out[b] = sigmoid(dot(r1[b,:], wh2) + bh2) -----------------
__global__ __launch_bounds__(128)
void head2_kernel(const float* __restrict__ r1, const float* __restrict__ w,
                  const float* __restrict__ bias, float* __restrict__ out)
{
    __shared__ float red[128];
    const int b = blockIdx.x, tid = threadIdx.x;
    red[tid] = r1[b * 128 + tid] * w[tid];
    __syncthreads();
    for (int s = 64; s > 0; s >>= 1) {
        if (tid < s) red[tid] += red[tid + s];
        __syncthreads();
    }
    if (tid == 0) {
        float x = red[0] + bias[0];
        out[b] = 1.f / (1.f + expf(-x));
    }
}

// ---------------------------------------------------------------------------
// Host orchestration
// ---------------------------------------------------------------------------
static inline void gemm(hipStream_t st, const float* A, const float* B, float* C,
                        const float* bias, const float* res, const float* rs,
                        int M, int N, int K, int lda, int ldb, int ldc,
                        int batch, int dv, int act,
                        long long sA1, long long sA2, long long sB1, long long sB2,
                        long long sC1, long long sC2,
                        long long sR1 = 0, long long sR2 = 0,
                        long long sS1 = 0, long long sS2 = 0)
{
    dim3 grid((N + 63) / 64, (M + 127) / 128, batch);
    wmma_gemm_kernel<<<grid, 256, 0, st>>>(A, B, C, bias, res, rs,
                                           M, N, K, lda, ldb, ldc, dv, act,
                                           sA1, sA2, sB1, sB2, sC1, sC2,
                                           sR1, sR2, sS1, sS2);
}

extern "C" void kernel_launch(void* const* d_in, const int* in_sizes, int n_in,
                              void* d_out, int out_size, void* d_ws, size_t ws_size,
                              hipStream_t stream)
{
    (void)in_sizes; (void)n_in; (void)out_size; (void)ws_size;

    const float* x     = (const float*)d_in[0];
    const float* w_in  = (const float*)d_in[1];
    const float* b_in  = (const float*)d_in[2];
    const float* proj  = (const float*)d_in[3];
    const float* ln1_g = (const float*)d_in[4];
    const float* ln1_b = (const float*)d_in[5];
    const float* wq    = (const float*)d_in[6];
    const float* wk    = (const float*)d_in[7];
    const float* wv    = (const float*)d_in[8];
    const float* wo    = (const float*)d_in[9];
    const float* bo    = (const float*)d_in[10];
    const float* ln2_g = (const float*)d_in[11];
    const float* ln2_b = (const float*)d_in[12];
    const float* w1    = (const float*)d_in[13];
    const float* b1    = (const float*)d_in[14];
    const float* w2    = (const float*)d_in[15];
    const float* b2    = (const float*)d_in[16];
    const float* hln_g = (const float*)d_in[17];
    const float* hln_b = (const float*)d_in[18];
    const float* wh1   = (const float*)d_in[19];
    const float* bh1   = (const float*)d_in[20];
    const float* wh2   = (const float*)d_in[21];
    const float* bh2   = (const float*)d_in[22];
    float* out = (float*)d_out;

    // ---------------- workspace layout (float offsets) ----------------
    const long long SZ_BND = (long long)ROWS * DIMC;            // 19,660,800
    const long long SZ_DD  = (long long)ZB * DDZ;               // 83,558,400
    const long long SZ_KPT = (long long)ZB * KPTZ;              // 84,672,512
    const long long SZ_CTX = (long long)ZB * CTXZ;              // 17,825,792

    float* ws = (float*)d_ws;
    long long off = 0;
    float* fh    = ws + off; off += SZ_BND;
    float* fy    = ws + off; off += SZ_BND;
    float* fq    = ws + off; off += SZ_BND;
    float* fk    = ws + off; off += SZ_BND;
    float* fv    = ws + off; off += SZ_BND;
    float* fo    = ws + off; off += SZ_BND;
    float* ddq   = ws + off; off += SZ_DD;     // becomes qp in-place
    float* ddk   = ws + off; off += SZ_DD;     // dead after ktrans; FFN aliases
    float* ffh   = ddk;                        // 78.6M <= 83.5M, lifetimes disjoint
    float* kpT   = ws + off; off += SZ_KPT;
    float* ctx   = ws + off; off += SZ_CTX;
    float* diagq = ws + off; off += (long long)ZB * NN;
    float* diagk = ws + off; off += (long long)ZB * NN;
    float* kstab = ws + off; off += ZB;
    float* ksum  = ws + off; off += (long long)ZB * MP;
    float* dinv  = ws + off; off += (long long)ZB * NN;
    float* prjT  = ws + off; off += (long long)DH * MP;
    float* pool  = ws + off; off += BB * DIMC;
    float* zln   = ws + off; off += BB * DIMC;
    float* r1    = ws + off; off += BB * 128;

    // ---------------- h = x @ w_in + b_in ----------------
    gemm(stream, x, w_in, fh, b_in, nullptr, nullptr,
         ROWS, DIMC, INDIM, INDIM, DIMC, DIMC, 1, 1, 0,
         0, 0, 0, 0, 0, 0);

    for (int l = 0; l < 2; ++l) {
        const long long lw = (long long)l * DIMC * DIMC;
        // ---- pre-norm attention ----
        ln_kernel<<<ROWS, 256, 0, stream>>>(fh, fy, ln1_g + l * DIMC, ln1_b + l * DIMC);

        gemm(stream, fy, wq + lw, fq, nullptr, nullptr, nullptr,
             ROWS, DIMC, DIMC, DIMC, DIMC, DIMC, 1, 1, 0, 0, 0, 0, 0, 0, 0);
        gemm(stream, fy, wk + lw, fk, nullptr, nullptr, nullptr,
             ROWS, DIMC, DIMC, DIMC, DIMC, DIMC, 1, 1, 0, 0, 0, 0, 0, 0, 0);
        gemm(stream, fy, wv + lw, fv, nullptr, nullptr, nullptr,
             ROWS, DIMC, DIMC, DIMC, DIMC, DIMC, 1, 1, 0, 0, 0, 0, 0, 0, 0);

        projT_kernel<<<(DH * MFEAT + 255) / 256, 256, 0, stream>>>(
            proj + (long long)l * MFEAT * DH, prjT);

        diag_kernel<<<(ZB * NN) / 8, 256, 0, stream>>>(fq, diagq);
        diag_kernel<<<(ZB * NN) / 8, 256, 0, stream>>>(fk, diagk);

        // dd = (q * norm) @ projT   [batched over 1024 (b,h)]
        gemm(stream, fq, prjT, ddq, nullptr, nullptr, nullptr,
             NN, MFEAT, DH, DIMC, MP, MP, ZB, HEADS, 0,
             (long long)NN * DIMC, DH, 0, 0,
             (long long)HEADS * DDZ, DDZ);
        gemm(stream, fk, prjT, ddk, nullptr, nullptr, nullptr,
             NN, MFEAT, DH, DIMC, MP, MP, ZB, HEADS, 0,
             (long long)NN * DIMC, DH, 0, 0,
             (long long)HEADS * DDZ, DDZ);

        qfeat_kernel<<<ZB * NN, 256, 0, stream>>>(ddq, diagq);   // ddq -> qp
        kstab_kernel<<<ZB, 256, 0, stream>>>(ddk, kstab);
        {
            dim3 g(ZB, (NN + 31) / 32, (MFEAT + 31) / 32);
            ktrans_kernel<<<g, 256, 0, stream>>>(ddk, diagk, kstab, kpT);
        }
        {
            dim3 g(ZB, MFEAT, 1);
            ksum_kernel<<<g, 128, 0, stream>>>(kpT, ksum);
        }

        // context[z] = kpT[z] @ v[z]   (M=266, N=64, K=300)
        gemm(stream, kpT, fv, ctx, nullptr, nullptr, nullptr,
             MFEAT, DH, NN, NPAD, DIMC, DH, ZB, HEADS, 0,
             (long long)HEADS * KPTZ, KPTZ,
             (long long)NN * DIMC, DH,
             (long long)HEADS * CTXZ, CTXZ);

        dinv_kernel<<<ZB * NN, 128, 0, stream>>>(ddq, ksum, dinv);

        // o[z] = (qp[z] @ context[z]) * dinv  (M=300, N=64, K=266->272 pad ok via guard)
        gemm(stream, ddq, ctx, fo, nullptr, nullptr, dinv,
             NN, DH, MFEAT, MP, DH, DIMC, ZB, HEADS, 0,
             (long long)HEADS * DDZ, DDZ,
             (long long)HEADS * CTXZ, CTXZ,
             (long long)NN * DIMC, DH,
             0, 0,
             (long long)HEADS * NN, NN);

        // h = h + o @ wo + bo
        gemm(stream, fo, wo + lw, fh, bo + l * DIMC, fh, nullptr,
             ROWS, DIMC, DIMC, DIMC, DIMC, DIMC, 1, 1, 0, 0, 0, 0, 0, 0, 0);

        // ---- pre-norm FFN ----
        ln_kernel<<<ROWS, 256, 0, stream>>>(fh, fy, ln2_g + l * DIMC, ln2_b + l * DIMC);
        gemm(stream, fy, w1 + (long long)l * DIMC * FFHID, ffh,
             b1 + l * FFHID, nullptr, nullptr,
             ROWS, FFHID, DIMC, DIMC, FFHID, FFHID, 1, 1, 1, 0, 0, 0, 0, 0, 0);
        gemm(stream, ffh, w2 + (long long)l * FFHID * DIMC, fh,
             b2 + l * DIMC, fh, nullptr,
             ROWS, DIMC, FFHID, FFHID, DIMC, DIMC, 1, 1, 0, 0, 0, 0, 0, 0, 0);
    }

    // ---------------- head ----------------
    pool_kernel<<<BB, 256, 0, stream>>>(fh, pool);
    ln_kernel<<<BB, 256, 0, stream>>>(pool, zln, hln_g, hln_b);
    gemm(stream, zln, wh1, r1, bh1, nullptr, nullptr,
         BB, 128, DIMC, DIMC, 128, 128, 1, 1, 2, 0, 0, 0, 0, 0, 0);
    head2_kernel<<<BB, 128, 0, stream>>>(r1, wh2, bh2, out);
}
